// PixelSnail_20804821582360
// MI455X (gfx1250) — matrix operand order repaired and checked by reading, not compiled, not run on hardware
//
#include <hip/hip_runtime.h>
#include <hip/hip_bf16.h>

// ---------------------------------------------------------------------------
// PixelSNAIL forward for MI455X (gfx1250, wave32).
// All channel contractions run through v_wmma_f32_16x16x32_bf16.
// Total ~20 GFLOP, ~22MB live data -> fits in 192MB L2; compute/latency bound.
// v2: native __bf16 converts (v_cvt_pk_bf16_f32) instead of bit-twiddling;
//     whole packed weight staged in LDS once (K now compile-time), so the
//     K loop runs barrier-free with unrolled WMMA chains.
// ---------------------------------------------------------------------------

typedef __attribute__((ext_vector_type(16))) __bf16 bf16x16;
typedef __attribute__((ext_vector_type(8)))  float  floatx8;

union AFrag { bf16x16 v; __bf16 e[16]; };
union BFrag { bf16x16 v; uint4 q[2]; };

#define NPIX 8192   // B*H*W = 8*32*32

__device__ __forceinline__ float elu1(float x) { return x > 0.f ? x : (__expf(x) - 1.f); }
__device__ __forceinline__ float sigm(float x) { return 1.f / (1.f + __expf(-x)); }

__device__ __forceinline__ unsigned packbf2(float lo, float hi) {
  union { __bf16 b[2]; unsigned u; } p;
  p.b[0] = (__bf16)lo; p.b[1] = (__bf16)hi;
  return p.u;
}

// ---------------------------------------------------------------------------
// Weight repack: original fp32 weights -> bf16 pairs laid out exactly as the
// per-lane WMMA B fragments, grouped per 32-wide K step:
//   dst[((ks*NT + nt)*32 + lane)*8 + j] = pack(W[k][col], W[k+1][col])
//   col = nt*16 + (lane&15), k = ks*32 + (lane>>4)*8 + (j>>2)*16 + 2*(j&3)
// ---------------------------------------------------------------------------
__global__ void repack_conv(const float* __restrict__ src, unsigned* __restrict__ dst,
                            int K, int N) {
  int idx = blockIdx.x * 256 + threadIdx.x;
  int total = (K * N) >> 1;
  if (idx >= total) return;
  int NT = N >> 4;
  int perks = NT * 256;
  int ks = idx / perks; int rem = idx % perks;
  int nt = rem >> 8;    int rem2 = rem & 255;
  int lane = rem2 >> 3; int j = rem2 & 7;
  int col = nt * 16 + (lane & 15);
  int k = ks * 32 + ((lane >> 4) << 3) + ((j >> 2) << 4) + 2 * (j & 3);
  // W[k][col] for conv2x2: k = tap*64 + ci; src layout (O, 64, 2, 2)
  float w0, w1;
  {
    int tap = k >> 6, ci = k & 63;
    w0 = src[(((col * 64 + ci) * 2 + (tap >> 1)) * 2 + (tap & 1))];
  }
  {
    int kk = k + 1;
    int tap = kk >> 6, ci = kk & 63;
    w1 = src[(((col * 64 + ci) * 2 + (tap >> 1)) * 2 + (tap & 1))];
  }
  dst[idx] = packbf2(w0, w1);
}

__global__ void repack_lin(const float* __restrict__ src, unsigned* __restrict__ dst,
                           int K, int N, int ldk) {
  int idx = blockIdx.x * 256 + threadIdx.x;
  int total = (K * N) >> 1;
  if (idx >= total) return;
  int NT = N >> 4;
  int perks = NT * 256;
  int ks = idx / perks; int rem = idx % perks;
  int nt = rem >> 8;    int rem2 = rem & 255;
  int lane = rem2 >> 3; int j = rem2 & 7;
  int col = nt * 16 + (lane & 15);
  int k = ks * 32 + ((lane >> 4) << 3) + ((j >> 2) << 4) + 2 * (j & 3);
  dst[idx] = packbf2(src[col * ldk + k], src[col * ldk + k + 1]);
}

// ---------------------------------------------------------------------------
// WMMA GEMM: Out(M=8192, N) = op(A)(M,K) * W(K,N) + bias, with fused pre/post.
//  PREOP: 0 none, 1 elu(a), 2 elu(a + a2)
//  IM2COL: A rows are pixels; K = tap*64 + ci over a 2x2/pad1/crop conv
//  POSTELU: elu on result; RES: add Rsrc[row][col]
// Block = 128 threads (4 waves); each wave owns a 16-row band, all N tiles.
// Entire packed weight (K*N/2 uints, <= 64KB) staged to LDS once; the K loop
// is compile-time unrolled and barrier-free.
// ---------------------------------------------------------------------------
template<int PREOP, bool IM2COL, int N, int K, bool POSTELU, bool RES>
__global__ __launch_bounds__(128) void gemm_wmma(
    const float* __restrict__ A, const float* __restrict__ A2,
    const unsigned* __restrict__ Wp, const float* __restrict__ bias,
    const float* __restrict__ Rsrc, float* __restrict__ Out)
{
  constexpr int NT = N / 16;
  constexpr int KSTEPS = K / 32;
  constexpr int WUINTS = (K * N) / 2;       // whole packed weight, in uints
  __shared__ __align__(16) unsigned bsh[WUINTS];

  const int tid  = threadIdx.x;
  const int wave = tid >> 5;
  const int lane = tid & 31;
  const int rowA = blockIdx.x * 64 + wave * 16 + (lane & 15);
  const int koff = (lane >> 4) << 3;   // lane<16: K 0..7,16..23 ; lane>=16: 8..15,24..31

  // stage the whole weight once (pure uint4 copies)
  {
    const uint4* wsrc = (const uint4*)Wp;
    uint4* wdst = (uint4*)bsh;
    for (int i = tid; i < WUINTS / 4; i += 128) wdst[i] = wsrc[i];
  }

  floatx8 acc[NT];
  #pragma unroll
  for (int t = 0; t < NT; ++t)
    #pragma unroll
    for (int r = 0; r < 8; ++r) acc[t][r] = 0.f;

  __syncthreads();

  #pragma unroll
  for (int ks = 0; ks < KSTEPS; ++ks) {
    float av[16];
    const int k0 = ks << 5;
    if constexpr (IM2COL) {
      const int kb  = k0 + koff;          // both 8-runs share one tap (taps at 64-mult)
      const int tap = kb >> 6, ci = kb & 63;
      const int b   = rowA >> 10, pix = rowA & 1023;
      const int y   = pix >> 5, x = pix & 31;
      const int yy  = y - 1 + (tap >> 1), xx = x - 1 + (tap & 1);
      if (yy >= 0 && xx >= 0 && xx < 32) {
        const float* rp = A + (((size_t)((b << 10) + (yy << 5) + xx)) << 6) + ci;
        const float4 v0 = ((const float4*)rp)[0];
        const float4 v1 = ((const float4*)rp)[1];
        const float4 v2 = ((const float4*)(rp + 16))[0];
        const float4 v3 = ((const float4*)(rp + 16))[1];
        av[0]=v0.x; av[1]=v0.y; av[2]=v0.z; av[3]=v0.w;
        av[4]=v1.x; av[5]=v1.y; av[6]=v1.z; av[7]=v1.w;
        av[8]=v2.x; av[9]=v2.y; av[10]=v2.z; av[11]=v2.w;
        av[12]=v3.x; av[13]=v3.y; av[14]=v3.z; av[15]=v3.w;
      } else {
        #pragma unroll
        for (int i = 0; i < 16; ++i) av[i] = 0.f;  // zero padding (elu(0)==0)
      }
    } else {
      const float* rp = A + (size_t)rowA * K + k0 + koff;
      const float4 v0 = ((const float4*)rp)[0];
      const float4 v1 = ((const float4*)rp)[1];
      const float4 v2 = ((const float4*)(rp + 16))[0];
      const float4 v3 = ((const float4*)(rp + 16))[1];
      av[0]=v0.x; av[1]=v0.y; av[2]=v0.z; av[3]=v0.w;
      av[4]=v1.x; av[5]=v1.y; av[6]=v1.z; av[7]=v1.w;
      av[8]=v2.x; av[9]=v2.y; av[10]=v2.z; av[11]=v2.w;
      av[12]=v3.x; av[13]=v3.y; av[14]=v3.z; av[15]=v3.w;
      if constexpr (PREOP == 2) {
        const float* rp2 = A2 + (size_t)rowA * K + k0 + koff;
        const float4 u0 = ((const float4*)rp2)[0];
        const float4 u1 = ((const float4*)rp2)[1];
        const float4 u2 = ((const float4*)(rp2 + 16))[0];
        const float4 u3 = ((const float4*)(rp2 + 16))[1];
        av[0]+=u0.x; av[1]+=u0.y; av[2]+=u0.z; av[3]+=u0.w;
        av[4]+=u1.x; av[5]+=u1.y; av[6]+=u1.z; av[7]+=u1.w;
        av[8]+=u2.x; av[9]+=u2.y; av[10]+=u2.z; av[11]+=u2.w;
        av[12]+=u3.x; av[13]+=u3.y; av[14]+=u3.z; av[15]+=u3.w;
      }
    }
    if constexpr (PREOP >= 1) {
      #pragma unroll
      for (int i = 0; i < 16; ++i) av[i] = elu1(av[i]);
    }
    AFrag af;
    #pragma unroll
    for (int i = 0; i < 16; ++i) af.e[i] = (__bf16)av[i];   // v_cvt_pk_bf16_f32

    #pragma unroll
    for (int t = 0; t < NT; ++t) {
      BFrag bf;
      const uint4* bp = (const uint4*)(bsh + ((ks * NT + t) * 32 + lane) * 8);
      bf.q[0] = bp[0]; bf.q[1] = bp[1];
      acc[t] = __builtin_amdgcn_wmma_f32_16x16x32_bf16(
          false, af.v, false, bf.v, (short)0, acc[t], false, false);
    }
  }

  // C/D layout: lanes 0-15 rows r=0..7, lanes 16-31 rows 8..15; col = lane&15
  const int rowC = blockIdx.x * 64 + wave * 16 + ((lane >> 4) << 3);
  const int colb = lane & 15;
  #pragma unroll
  for (int t = 0; t < NT; ++t) {
    const int col = t * 16 + colb;
    const float bv = bias[col];
    #pragma unroll
    for (int r = 0; r < 8; ++r) {
      const int row = rowC + r;
      float v = acc[t][r] + bv;
      if constexpr (POSTELU) v = elu1(v);
      if constexpr (RES) v += Rsrc[(size_t)row * N + col];
      Out[(size_t)row * N + col] = v;
    }
  }
}

// ---------------------------------------------------------------------------
// Stem: masked 3x3 conv (type-A mask keeps taps (0,0),(0,1),(0,2),(1,0)),
// CH=1 -> 64, writes NHWC fp32 [8192][64].
// ---------------------------------------------------------------------------
__global__ __launch_bounds__(256) void stem_kernel(
    const float* __restrict__ x, const float* __restrict__ w,
    const float* __restrict__ b, float* __restrict__ out)
{
  int t = blockIdx.x * 256 + threadIdx.x;
  if (t >= NPIX * 64) return;
  int o = t & 63, n = t >> 6;
  int bb = n >> 10, pix = n & 1023, y = pix >> 5, xx = pix & 31;
  const float* xi = x + (bb << 10);
  const float* wo = w + o * 9;       // (1,3,3)
  float acc = b[o];
  if (y >= 1) {
    int yb = (y - 1) << 5;
    if (xx >= 1)  acc += wo[0] * xi[yb + xx - 1];
    acc += wo[1] * xi[yb + xx];
    if (xx <= 30) acc += wo[2] * xi[yb + xx + 1];
  }
  if (xx >= 1) acc += wo[3] * xi[(y << 5) + xx - 1];
  out[(size_t)n * 64 + o] = acc;
}

// gated residual: conv += t2[:, :64] * sigmoid(t2[:, 64:])
__global__ __launch_bounds__(256) void gate_kernel(
    float* __restrict__ conv, const float* __restrict__ t2)
{
  int t = blockIdx.x * 256 + threadIdx.x;
  if (t >= NPIX * 64) return;
  int c = t & 63, n = t >> 6;
  conv[t] += t2[(size_t)n * 128 + c] * sigm(t2[(size_t)n * 128 + 64 + c]);
}

// add positional-encoding channels' contribution to qkv (in-channels 64,65)
__global__ __launch_bounds__(256) void qkvpos_kernel(
    float* __restrict__ qkv, const float* __restrict__ qw /* (64,66) slice */)
{
  int t = blockIdx.x * 256 + threadIdx.x;
  if (t >= NPIX * 64) return;
  int o = t & 63, n = t >> 6;
  int pix = n & 1023, y = pix >> 5, x = pix & 31;
  float py = y * (1.f / 32.f) - 0.5f;
  float px = x * (1.f / 32.f) - 0.5f;
  qkv[t] += qw[o * 66 + 64] * py + qw[o * 66 + 65] * px;
}

// ---------------------------------------------------------------------------
// Causal attention, flash-style online softmax. qk per-head dim = 4,
// v per-head dim = 8 -> VALU, K/V staged in LDS (48KB << 320KB WGP LDS).
// grid = (B*HEAD, n/256), block = 256 (one thread per query).
// qkv channels: q = h*4.., k = 16+h*4.., v = 32+h*8.. ; scale = 0.5
// ---------------------------------------------------------------------------
__global__ __launch_bounds__(256) void attn_kernel(
    const float* __restrict__ qkv, float* __restrict__ obuf)
{
  __shared__ float ksh[1024 * 4];
  __shared__ float vsh[1024 * 8];
  int bh = blockIdx.x; int b = bh >> 2, h = bh & 3;
  int tid = threadIdx.x;
  for (int i = tid; i < 1024; i += 256) {
    const float* base = qkv + ((size_t)((b << 10) + i)) * 64;
    #pragma unroll
    for (int d = 0; d < 4; ++d) ksh[i * 4 + d] = base[16 + h * 4 + d];
    #pragma unroll
    for (int d = 0; d < 8; ++d) vsh[i * 8 + d] = base[32 + h * 8 + d];
  }
  __syncthreads();

  int qn = blockIdx.y * 256 + tid;
  const float* qb = qkv + ((size_t)((b << 10) + qn)) * 64 + h * 4;
  float q0 = qb[0] * 0.5f, q1 = qb[1] * 0.5f, q2 = qb[2] * 0.5f, q3 = qb[3] * 0.5f;

  float m = -3.0e38f, l = 0.f, acc[8];
  #pragma unroll
  for (int d = 0; d < 8; ++d) acc[d] = 0.f;

  for (int j = 0; j <= qn; ++j) {      // causal: keys <= query
    const float* kp = ksh + j * 4;
    float s = q0 * kp[0] + q1 * kp[1] + q2 * kp[2] + q3 * kp[3];
    float p;
    if (s > m) {
      float f = __expf(m - s);
      l *= f;
      #pragma unroll
      for (int d = 0; d < 8; ++d) acc[d] *= f;
      m = s; p = 1.f;
    } else {
      p = __expf(s - m);
    }
    l += p;
    const float* vp = vsh + j * 8;
    #pragma unroll
    for (int d = 0; d < 8; ++d) acc[d] += p * vp[d];
  }
  float inv = 1.f / l;
  float* op = obuf + ((size_t)((b << 10) + qn)) * 32 + h * 8;
  #pragma unroll
  for (int d = 0; d < 8; ++d) op[d] = acc[d] * inv;
}

// final fc: logits[n] = fc_b + sum_c fc_w[c] * elu(act[n][c]); output NCHW==n
__global__ __launch_bounds__(256) void fc_kernel(
    const float* __restrict__ act, const float* __restrict__ fw,
    const float* __restrict__ fb, float* __restrict__ out)
{
  int n = blockIdx.x * 256 + threadIdx.x;
  if (n >= NPIX) return;
  const float* a = act + (size_t)n * 64;
  float acc = fb[0];
  #pragma unroll
  for (int c = 0; c < 64; ++c) acc += fw[c] * elu1(a[c]);
  out[n] = acc;
}

// ---------------------------------------------------------------------------
extern "C" void kernel_launch(void* const* d_in, const int* in_sizes, int n_in,
                              void* d_out, int out_size, void* d_ws, size_t ws_size,
                              hipStream_t stream)
{
  const float* x      = (const float*)d_in[0];
  const float* stem_w = (const float*)d_in[1];
  const float* stem_b = (const float*)d_in[2];
  const float* bb1_w  = (const float*)d_in[3];
  const float* bb1_b  = (const float*)d_in[4];
  const float* bb2_w  = (const float*)d_in[5];
  const float* bb2_b  = (const float*)d_in[6];
  const float* qkv_w  = (const float*)d_in[7];
  const float* qkv_b  = (const float*)d_in[8];
  const float* proj_w = (const float*)d_in[9];
  const float* proj_b = (const float*)d_in[10];
  const float* outc_w = (const float*)d_in[11];
  const float* outc_b = (const float*)d_in[12];
  const float* outa_w = (const float*)d_in[13];
  const float* outa_b = (const float*)d_in[14];
  const float* outp_w = (const float*)d_in[15];
  const float* outp_b = (const float*)d_in[16];
  const float* fc_w   = (const float*)d_in[17];
  const float* fc_b   = (const float*)d_in[18];

  char* ws = (char*)d_ws;
  size_t cur = 0;
  auto alloc = [&](size_t bytes) -> void* {
    void* p = ws + cur;
    cur = (cur + bytes + 255) & ~(size_t)255;
    return p;
  };
  const size_t N64B = (size_t)NPIX * 64 * 4;
  float* bufOut  = (float*)alloc(N64B);                  // residual 'out'
  float* bufConv = (float*)alloc(N64B);                  // 'conv'
  float* bufT1   = (float*)alloc(N64B);                  // block temp
  float* bufT2   = (float*)alloc((size_t)NPIX * 128 * 4);// gated 128-ch temp
  float* bufQkv  = (float*)alloc(N64B);
  float* bufO    = (float*)alloc((size_t)NPIX * 32 * 4); // attention output
  float* bufAttn = (float*)alloc(N64B);                  // attn after proj
  float* bufCO   = (float*)alloc(N64B);                  // outc result
  float* bufAO   = (float*)alloc(N64B);                  // outa result
  unsigned* pk_bb1  = (unsigned*)alloc((size_t)16 * 8192  * 4);
  unsigned* pk_bb2  = (unsigned*)alloc((size_t)16 * 16384 * 4);
  unsigned* pk_qkv  = (unsigned*)alloc((size_t)8  * 2048  * 4);
  unsigned* pk_proj = (unsigned*)alloc((size_t)8  * 1024  * 4);
  unsigned* pk_outc = (unsigned*)alloc((size_t)8  * 2048  * 4);
  unsigned* pk_outa = (unsigned*)alloc((size_t)8  * 2048  * 4);
  unsigned* pk_outp = (unsigned*)alloc((size_t)8  * 2048  * 4);

  // ---- repack all weights into WMMA B-fragment bf16 layout --------------
  for (int sl = 0; sl < 16; ++sl) {
    repack_conv<<<8192 / 256, 256, 0, stream>>>(
        bb1_w + (size_t)sl * 64 * 64 * 4, pk_bb1 + (size_t)sl * 8192, 256, 64);
    repack_conv<<<16384 / 256, 256, 0, stream>>>(
        bb2_w + (size_t)sl * 128 * 64 * 4, pk_bb2 + (size_t)sl * 16384, 256, 128);
  }
  for (int s = 0; s < 8; ++s) {
    repack_lin<<<2048 / 256, 256, 0, stream>>>(
        qkv_w + (size_t)s * 64 * 66, pk_qkv + (size_t)s * 2048, 64, 64, 66);
    repack_lin<<<1024 / 256, 256, 0, stream>>>(
        proj_w + (size_t)s * 64 * 32, pk_proj + (size_t)s * 1024, 32, 64, 32);
    repack_lin<<<2048 / 256, 256, 0, stream>>>(
        outc_w + (size_t)s * 64 * 64, pk_outc + (size_t)s * 2048, 64, 64, 64);
    repack_lin<<<2048 / 256, 256, 0, stream>>>(
        outa_w + (size_t)s * 64 * 64, pk_outa + (size_t)s * 2048, 64, 64, 64);
    repack_lin<<<2048 / 256, 256, 0, stream>>>(
        outp_w + (size_t)s * 64 * 64, pk_outp + (size_t)s * 2048, 64, 64, 64);
  }

  // ---- stem --------------------------------------------------------------
  stem_kernel<<<(NPIX * 64) / 256, 256, 0, stream>>>(x, stem_w, stem_b, bufOut);

  // ---- stages ------------------------------------------------------------
  for (int s = 0; s < 8; ++s) {
    hipMemcpyAsync(bufConv, bufOut, N64B, hipMemcpyDeviceToDevice, stream);

    for (int l = 0; l < 2; ++l) {
      int sl = s * 2 + l;
      gemm_wmma<1, true, 64, 256, false, false><<<128, 128, 0, stream>>>(
          bufConv, nullptr, pk_bb1 + (size_t)sl * 8192, bb1_b + sl * 64,
          nullptr, bufT1);
      gemm_wmma<1, true, 128, 256, false, false><<<128, 128, 0, stream>>>(
          bufT1, nullptr, pk_bb2 + (size_t)sl * 16384, bb2_b + sl * 128,
          nullptr, bufT2);
      gate_kernel<<<(NPIX * 64) / 256, 256, 0, stream>>>(bufConv, bufT2);
    }

    gemm_wmma<0, false, 64, 64, false, false><<<128, 128, 0, stream>>>(
        bufConv, nullptr, pk_qkv + (size_t)s * 2048, qkv_b + s * 64,
        nullptr, bufQkv);
    qkvpos_kernel<<<(NPIX * 64) / 256, 256, 0, stream>>>(
        bufQkv, qkv_w + (size_t)s * 64 * 66);
    attn_kernel<<<dim3(32, 4), 256, 0, stream>>>(bufQkv, bufO);
    gemm_wmma<0, false, 64, 32, false, false><<<128, 128, 0, stream>>>(
        bufO, nullptr, pk_proj + (size_t)s * 1024, proj_b + s * 64,
        nullptr, bufAttn);

    gemm_wmma<1, false, 64, 64, true, false><<<128, 128, 0, stream>>>(
        bufConv, nullptr, pk_outc + (size_t)s * 2048, outc_b + s * 64,
        nullptr, bufCO);
    gemm_wmma<1, false, 64, 64, true, false><<<128, 128, 0, stream>>>(
        bufAttn, nullptr, pk_outa + (size_t)s * 2048, outa_b + s * 64,
        nullptr, bufAO);
    gemm_wmma<2, false, 64, 64, true, true><<<128, 128, 0, stream>>>(
        bufCO, bufAO, pk_outp + (size_t)s * 2048, outp_b + s * 64,
        bufOut, bufOut);
  }

  // ---- final fc ----------------------------------------------------------
  fc_kernel<<<NPIX / 256, 256, 0, stream>>>(bufOut, fc_w, fc_b, (float*)d_out);
}
